// Attention_15745350107626
// MI455X (gfx1250) — matrix operand order repaired and checked by reading, not compiled, not run on hardware
//
#include <hip/hip_runtime.h>
#include <hip/hip_bf16.h>

#define BATCH 2
#define NSEQ 2048
#define DIMC 1024
#define NH 16
#define HD 64

typedef __attribute__((ext_vector_type(16))) __bf16 v16bf;
typedef __attribute__((ext_vector_type(8)))  __bf16 v8bf;
typedef __attribute__((ext_vector_type(8)))  float  v8f;

static __device__ inline __bf16 f2bf(float f) {
    unsigned u = __builtin_bit_cast(unsigned, f);
    u += 0x7FFFu + ((u >> 16) & 1u);           // round-to-nearest-even
    unsigned short s = (unsigned short)(u >> 16);
    return __builtin_bit_cast(__bf16, s);
}

static __device__ inline v16bf cat8(v8bf lo, v8bf hi) {
    return __builtin_shufflevector(lo, hi, 0,1,2,3,4,5,6,7,8,9,10,11,12,13,14,15);
}

static __device__ inline v8f wmma_bf16(v16bf a, v16bf b, v8f c) {
    return __builtin_amdgcn_wmma_f32_16x16x32_bf16(
        /*neg_a=*/false, a, /*neg_b=*/false, b,
        /*c_mod=*/(short)0, c, /*reuse_a=*/false, /*reuse_b=*/false);
}

// ---------------- elementwise f32 -> bf16 cast ----------------
__global__ void cast_f32_bf16(const float* __restrict__ in, __bf16* __restrict__ out, int n) {
    int i = blockIdx.x * blockDim.x + threadIdx.x;
    if (i < n) out[i] = f2bf(in[i]);
}

// ---------------- generic bf16 GEMM: C(MxN,f32) = A(MxK) @ B(KxN) ----------------
// A row-major bf16, B row-major bf16. One wave computes a 16x64 C tile
// (4 f32 accumulators) so each A fragment feeds 4 WMMAs. 4 waves/block on N
// -> a block covers 16x256 of C. Weights (B) stream through L2 (192 MB).
// All fragment loads for a K-step are issued as one clause before the WMMAs
// so waits can be staged; prefetch is unconditional (speculative, droppable).
__global__ void gemm_bf16(const __bf16* __restrict__ A, const __bf16* __restrict__ Bm,
                          float* __restrict__ C, int M, int N, int K) {
    int w = threadIdx.x >> 5, lane = threadIdx.x & 31;
    int ngrp = blockIdx.x * 4 + w;          // 64-column group
    int mtile = blockIdx.y;
    int row = mtile * 16 + (lane & 15);
    int k0 = (lane >> 4) * 8;
    const __bf16* arow = A + (size_t)row * K;

    v8f acc[4] = {};
    for (int kk = 0; kk < K; kk += 32) {
        const __bf16* brow = Bm + (size_t)(kk + lane) * N + ngrp * 64;
        // unconditional speculative prefetch of the next K-step
        __builtin_prefetch(arow + kk + 32 + k0, 0, 1);
        __builtin_prefetch(brow + (size_t)32 * N, 0, 1);

        v8bf alo = *(const v8bf*)(arow + kk + k0);
        v8bf ahi = *(const v8bf*)(arow + kk + k0 + 16);
        v16bf afrag = cat8(alo, ahi);
        v16bf bfr[4];
#pragma unroll
        for (int t = 0; t < 4; t++)
            for (int s = 0; s < 16; s++)
                bfr[t][s] = Bm[(size_t)(kk + (s < 8 ? k0 + s : 16 + k0 + s - 8)) * N + ngrp * 64 + t * 16 + (lane & 15)];
#pragma unroll
        for (int t = 0; t < 4; t++)
            acc[t] = wmma_bf16(afrag, bfr[t], acc[t]);
    }
    float* cbase = C + (size_t)(mtile * 16 + 8 * (lane >> 4)) * N + ngrp * 64 + (lane & 15);
#pragma unroll
    for (int t = 0; t < 4; t++)
#pragma unroll
        for (int r = 0; r < 8; r++)
            cbase[(size_t)r * N + t * 16] = acc[t][r];
}

// ---------------- per-head RMSNorm + RoPE ----------------
// X: (b, seq, ldx) f32 projection output; head column base = h*HD.
// transposed==0 -> out (b*h, seq, HD) bf16 (Q / A-operand layout)
// transposed==1 -> out (b*h, HD, seq) bf16 (K / B-operand layout, hd-major)
__global__ void norm_rope(const float* __restrict__ X, int ldx,
                          const float* __restrict__ pos, const float* __restrict__ w,
                          __bf16* __restrict__ out, int transposed) {
    int gw = (blockIdx.x * blockDim.x + threadIdx.x) >> 5;
    int lane = threadIdx.x & 31;
    int h = gw % NH;
    int t = gw / NH;
    int n = t % NSEQ;
    int b = t / NSEQ;
    const float* row = X + (size_t)(b * NSEQ + n) * ldx + h * HD;
    float x1 = row[lane], x2 = row[lane + 32];
    float ss = x1 * x1 + x2 * x2;
#pragma unroll
    for (int m = 16; m >= 1; m >>= 1) ss += __shfl_xor(ss, m, 32);
    float scale = rsqrtf(ss * (1.0f / 64.0f) + 1.1920929e-07f);
    float y1 = x1 * scale * w[lane];
    float y2 = x2 * scale * w[lane + 32];
    float p = pos[b * NSEQ + n];
    float inv_freq = powf(10000.0f, -(float)lane * (1.0f / 32.0f));
    float ang = p * inv_freq;
    float cs = cosf(ang), sn = sinf(ang);
    float o1 = y1 * cs - y2 * sn;
    float o2 = y1 * sn + y2 * cs;
    int bh = b * NH + h;
    if (!transposed) {
        __bf16* q = out + ((size_t)bh * NSEQ + n) * HD;
        q[lane] = f2bf(o1);
        q[lane + 32] = f2bf(o2);
    } else {
        __bf16* k = out + (size_t)bh * HD * NSEQ;
        k[(size_t)lane * NSEQ + n] = f2bf(o1);
        k[(size_t)(lane + 32) * NSEQ + n] = f2bf(o2);
    }
}

// ---------------- extract V from fused KV and cast: (b*h, m, HD) bf16 ----------------
__global__ void extract_v(const float* __restrict__ KV, __bf16* __restrict__ vout) {
    int i = blockIdx.x * blockDim.x + threadIdx.x;
    if (i >= BATCH * NSEQ * DIMC) return;
    int d = i % HD;
    int t = i / HD;
    int h = t % NH;
    int t2 = t / NH;
    int mrow = t2 % NSEQ;
    int b = t2 / NSEQ;
    float v = KV[(size_t)(b * NSEQ + mrow) * (2 * DIMC) + DIMC + h * HD + d];
    vout[(((size_t)(b * NH + h)) * NSEQ + mrow) * HD + d] = f2bf(v);
}

// ---------------- flash attention ----------------
// Q:(bh,n,HD) bf16, Kt:(bh,HD,m) bf16, V:(bh,m,HD) bf16 -> Of:(b,n,DIM) f32
// One wave per 16-query tile; 32-key steps; online softmax; P bounced via LDS
// to convert C-layout -> A-fragment layout for the PV WMMA.
__global__ void flash_attn(const __bf16* __restrict__ Q, const __bf16* __restrict__ Kt,
                           const __bf16* __restrict__ V, float* __restrict__ Of) {
    __shared__ __bf16 Pt[4][16][32];
    int w = threadIdx.x >> 5, lane = threadIdx.x & 31;
    int tile = blockIdx.x * 4 + w;
    int qblk = tile % (NSEQ / 16);
    int bh = tile / (NSEQ / 16);
    const __bf16* q = Q + ((size_t)bh * NSEQ + qblk * 16) * HD;
    const __bf16* kt = Kt + (size_t)bh * HD * NSEQ;
    const __bf16* vp = V + (size_t)bh * NSEQ * HD;
    int mrow = lane & 15, k0 = (lane >> 4) * 8;

    v16bf qf[2];
#pragma unroll
    for (int kc = 0; kc < 2; kc++) {
        v8bf lo = *(const v8bf*)(q + (size_t)mrow * HD + kc * 32 + k0);
        v8bf hi = *(const v8bf*)(q + (size_t)mrow * HD + kc * 32 + k0 + 16);
        qf[kc] = cat8(lo, hi);
    }

    v8f o[4] = {};
    float mr[8], lr[8];
#pragma unroll
    for (int r = 0; r < 8; r++) { mr[r] = -1.0e30f; lr[r] = 0.0f; }

    for (int j = 0; j < NSEQ / 32; j++) {
        // unconditional speculative prefetch of next key/value block
        __builtin_prefetch(kt + (size_t)lane * NSEQ + (j + 1) * 32, 0, 1);
        __builtin_prefetch(vp + (size_t)((j + 1) * 32 + lane) * HD, 0, 1);

        // S = Q @ K^T : load all K fragments first, then 4 WMMAs
        v16bf kf[4];
#pragma unroll
        for (int kc = 0; kc < 2; kc++) {
            for (int s = 0; s < 16; s++) {
                int kr = s < 8 ? k0 + s : 16 + k0 + s - 8;
                kf[kc * 2 + 0][s] = kt[(size_t)(kc * 32 + kr) * NSEQ + j * 32 + (lane & 15)];
                kf[kc * 2 + 1][s] = kt[(size_t)(kc * 32 + kr) * NSEQ + j * 32 + 16 + (lane & 15)];
            }
        }
        v8f c0 = {}, c1 = {};
        c0 = wmma_bf16(qf[0], kf[0], c0);
        c1 = wmma_bf16(qf[0], kf[1], c1);
        c0 = wmma_bf16(qf[1], kf[2], c0);
        c1 = wmma_bf16(qf[1], kf[3], c1);

#pragma unroll
        for (int r = 0; r < 8; r++) {
            float s0 = c0[r] * 0.125f;   // 1/sqrt(64)
            float s1 = c1[r] * 0.125f;
            float t = fmaxf(s0, s1);
#pragma unroll
            for (int msk = 8; msk >= 1; msk >>= 1) t = fmaxf(t, __shfl_xor(t, msk, 16));
            float nm = fmaxf(mr[r], t);
            float p0 = __expf(s0 - nm);
            float p1 = __expf(s1 - nm);
            float rs = p0 + p1;
#pragma unroll
            for (int msk = 8; msk >= 1; msk >>= 1) rs += __shfl_xor(rs, msk, 16);
            float alpha = __expf(mr[r] - nm);
            lr[r] = lr[r] * alpha + rs;
            mr[r] = nm;
#pragma unroll
            for (int tt = 0; tt < 4; tt++) o[tt][r] *= alpha;
            int prow = r + 8 * (lane >> 4);
            Pt[w][prow][lane & 15] = f2bf(p0);
            Pt[w][prow][16 + (lane & 15)] = f2bf(p1);
        }
        __syncthreads();
        v8bf plo = *(const v8bf*)&Pt[w][mrow][k0];
        v8bf phi = *(const v8bf*)&Pt[w][mrow][k0 + 16];
        v16bf pf = cat8(plo, phi);

        // O += P @ V : load all V fragments first, then 4 WMMAs
        v16bf vf[4];
#pragma unroll
        for (int tt = 0; tt < 4; tt++)
            for (int s = 0; s < 16; s++)
                vf[tt][s] = vp[(size_t)(j * 32 + (s < 8 ? k0 + s : 16 + k0 + s - 8)) * HD + tt * 16 + (lane & 15)];
#pragma unroll
        for (int tt = 0; tt < 4; tt++)
            o[tt] = wmma_bf16(pf, vf[tt], o[tt]);
        __syncthreads();
    }

    int b = bh / NH, h = bh % NH;
#pragma unroll
    for (int tt = 0; tt < 4; tt++) {
#pragma unroll
        for (int r = 0; r < 8; r++) {
            int nrow = qblk * 16 + r + 8 * (lane >> 4);
            Of[(size_t)(b * NSEQ + nrow) * DIMC + h * HD + tt * 16 + (lane & 15)] =
                o[tt][r] / lr[r];
        }
    }
}

extern "C" void kernel_launch(void* const* d_in, const int* in_sizes, int n_in,
                              void* d_out, int out_size, void* d_ws, size_t ws_size,
                              hipStream_t stream) {
    const float* tgt     = (const float*)d_in[0];
    const float* src     = (const float*)d_in[1];
    const float* tgt_pos = (const float*)d_in[2];
    const float* src_pos = (const float*)d_in[3];
    const float* Wq      = (const float*)d_in[4];
    const float* Wkv     = (const float*)d_in[5];
    const float* Wo      = (const float*)d_in[6];
    const float* qw      = (const float*)d_in[7];
    const float* kw      = (const float*)d_in[8];
    float* out = (float*)d_out;

    const size_t ROWS = (size_t)BATCH * NSEQ;       // 4096
    char* base = (char*)d_ws;
    size_t off = 0;
    auto carve = [&](size_t bytes) {
        char* p = base + off;
        off += (bytes + 255) & ~(size_t)255;
        return p;
    };
    __bf16* tgt_bf  = (__bf16*)carve(ROWS * DIMC * 2);
    __bf16* src_bf  = (__bf16*)carve(ROWS * DIMC * 2);
    __bf16* wq_bf   = (__bf16*)carve((size_t)DIMC * DIMC * 2);
    __bf16* wkv_bf  = (__bf16*)carve((size_t)DIMC * 2 * DIMC * 2);
    __bf16* wo_bf   = (__bf16*)carve((size_t)DIMC * DIMC * 2);
    float*  Qf      = (float*)carve(ROWS * DIMC * 4);
    float*  KVf     = (float*)carve(ROWS * 2 * DIMC * 4);
    __bf16* q_bf    = (__bf16*)carve(ROWS * DIMC * 2);   // (bh, n, HD)
    __bf16* kt_bf   = (__bf16*)carve(ROWS * DIMC * 2);   // (bh, HD, m)
    __bf16* v_bf    = (__bf16*)carve(ROWS * DIMC * 2);   // (bh, m, HD)
    float*  attnf   = (float*)carve(ROWS * DIMC * 4);    // (b, n, DIM)
    __bf16* attn_bf = (__bf16*)carve(ROWS * DIMC * 2);
    (void)ws_size; (void)in_sizes; (void)n_in; (void)out_size;

    auto cast = [&](const float* in, __bf16* o, size_t n) {
        cast_f32_bf16<<<dim3((unsigned)((n + 255) / 256)), dim3(256), 0, stream>>>(in, o, (int)n);
    };

    // 1) casts
    cast(tgt, tgt_bf, ROWS * DIMC);
    cast(src, src_bf, ROWS * DIMC);
    cast(Wq,  wq_bf,  (size_t)DIMC * DIMC);
    cast(Wkv, wkv_bf, (size_t)DIMC * 2 * DIMC);
    cast(Wo,  wo_bf,  (size_t)DIMC * DIMC);

    // 2) projections: Q = tgt@Wq, KV = src@Wkv  (16x64 tile per wave, 16x256 per block)
    gemm_bf16<<<dim3(DIMC / 256, (unsigned)(ROWS / 16)), dim3(128), 0, stream>>>(
        tgt_bf, wq_bf, Qf, (int)ROWS, DIMC, DIMC);
    gemm_bf16<<<dim3(2 * DIMC / 256, (unsigned)(ROWS / 16)), dim3(128), 0, stream>>>(
        src_bf, wkv_bf, KVf, (int)ROWS, 2 * DIMC, DIMC);

    // 3) per-head RMSNorm + RoPE (q row-major, k transposed per head)
    {
        unsigned waves = (unsigned)(BATCH * NSEQ * NH);
        norm_rope<<<dim3(waves / 4), dim3(128), 0, stream>>>(Qf, DIMC, tgt_pos, qw, q_bf, 0);
        norm_rope<<<dim3(waves / 4), dim3(128), 0, stream>>>(KVf, 2 * DIMC, src_pos, kw, kt_bf, 1);
    }

    // 4) V extract + cast
    {
        size_t n = ROWS * DIMC;
        extract_v<<<dim3((unsigned)((n + 255) / 256)), dim3(256), 0, stream>>>(KVf, v_bf);
    }

    // 5) flash attention
    {
        unsigned tiles = (unsigned)(BATCH * NH * (NSEQ / 16));  // 4096 waves
        flash_attn<<<dim3(tiles / 4), dim3(128), 0, stream>>>(q_bf, kt_bf, v_bf, attnf);
    }

    // 6) output projection into d_out
    cast(attnf, attn_bf, ROWS * DIMC);
    gemm_bf16<<<dim3(DIMC / 256, (unsigned)(ROWS / 16)), dim3(128), 0, stream>>>(
        attn_bf, wo_bf, out, (int)ROWS, DIMC, DIMC);
}
